// MultiheadAttention_24996709663149
// MI455X (gfx1250) — matrix operand order repaired and checked by reading, not compiled, not run on hardware
//
#include <hip/hip_runtime.h>
#include <hip/hip_bf16.h>

// ---------------------------------------------------------------------------
// MHA forward for MI455X (gfx1250, wave32, WMMA).
//  d_out = [ out f32 (2*2048*1024) | attn f32 (2*16*2048*2048) ]
// All matmuls via v_wmma_f32_16x16x32_bf16 (f32 accumulate).
// ---------------------------------------------------------------------------

#define BD   2          // batch
#define SS   2048       // sequence
#define DM   1024       // d_model
#define NH   16         // heads
#define DH   64         // head dim
#define MROWS (BD*SS)   // 4096
#define NEGC (-1.0e9f)
#define ISC  (0.03125f) // 1/sqrt(1024)

typedef __attribute__((ext_vector_type(16))) __bf16 v16bf;
typedef __attribute__((ext_vector_type(8)))  float  v8f;

static __device__ __forceinline__ v8f wmma_bf16(v16bf a, v16bf b, v8f c) {
  return __builtin_amdgcn_wmma_f32_16x16x32_bf16(false, a, false, b, (short)0, c,
                                                 false, false);
}

static __device__ __forceinline__ unsigned short f32_to_bf16_bits(float f) {
  union { float f; unsigned int u; } cv; cv.f = f;
  unsigned int u = cv.u;
  u += 0x7FFFu + ((u >> 16) & 1u);   // round to nearest even
  return (unsigned short)(u >> 16);
}

// A fragment (16x32 bf16), row-major source, p points at (row0, k0).
// Layout (ISA 7.12.2): m = lane&15 ; k = (v/4)*16 + (lane>>4)*8 + (v&3)*2.
static __device__ __forceinline__ v16bf load_a_bf16(
    const unsigned short* __restrict__ p, int ld, int lane) {
  int m = lane & 15, h = lane >> 4;
  const unsigned short* base = p + m * ld;
  v16bf a;
#pragma unroll
  for (int v = 0; v < 8; ++v) {
    int k = ((v >> 2) << 4) + (h << 3) + ((v & 3) << 1);
    unsigned int pair = *reinterpret_cast<const unsigned int*>(base + k);
    union { unsigned int u; __bf16 x[2]; } cv; cv.u = pair;
    a[2 * v] = cv.x[0]; a[2 * v + 1] = cv.x[1];
  }
  return a;
}

// A fragment from an f32 source (global or LDS), converted to bf16 on fill.
static __device__ __forceinline__ v16bf load_a_f32(
    const float* __restrict__ p, int ld, int lane) {
  int m = lane & 15, h = lane >> 4;
  const float* base = p + m * ld;
  v16bf a;
#pragma unroll
  for (int v = 0; v < 8; ++v) {
    int k = ((v >> 2) << 4) + (h << 3) + ((v & 3) << 1);
    float2 f = *reinterpret_cast<const float2*>(base + k);
    a[2 * v] = (__bf16)f.x; a[2 * v + 1] = (__bf16)f.y;
  }
  return a;
}

// B fragment (32x16 bf16) from an N-major source (each n-row contiguous in k).
// Layout: n = lane&15 ; k = (v/4)*16 + (lane>>4)*8 + (v&3)*2 (matches A).
static __device__ __forceinline__ v16bf load_b_bf16(
    const unsigned short* __restrict__ p, int ld, int lane) {
  int n = lane & 15, h = lane >> 4;
  const unsigned short* base = p + n * ld;
  v16bf b;
#pragma unroll
  for (int v = 0; v < 8; ++v) {
    int k = ((v >> 2) << 4) + (h << 3) + ((v & 3) << 1);
    unsigned int pair = *reinterpret_cast<const unsigned int*>(base + k);
    union { unsigned int u; __bf16 x[2]; } cv; cv.u = pair;
    b[2 * v] = cv.x[0]; b[2 * v + 1] = cv.x[1];
  }
  return b;
}

// ---------------------------------------------------------------------------
// Kernel 1: W (f32, row-major KxN) -> WT (bf16, N-major NxK)
// ---------------------------------------------------------------------------
__global__ void transpose_cvt_kernel(const float* __restrict__ w,
                                     unsigned short* __restrict__ wt) {
  __shared__ float tile[32][33];
  int bx = blockIdx.x * 32;  // column tile (n)
  int by = blockIdx.y * 32;  // row tile (k)
  int tx = threadIdx.x, ty = threadIdx.y;   // 32 x 8
#pragma unroll
  for (int i = 0; i < 32; i += 8)
    tile[ty + i][tx] = w[(by + ty + i) * DM + bx + tx];
  __syncthreads();
#pragma unroll
  for (int i = 0; i < 32; i += 8)
    wt[(bx + ty + i) * DM + by + tx] = f32_to_bf16_bits(tile[tx][ty + i]);
}

// ---------------------------------------------------------------------------
// Kernel 2: GEMM  Y[4096x1024] = X[4096x1024] . W + bias
//   mode 0: X=f32 q,  store bf16 head-split   qh[(b*16+h)*S + s][64]
//   mode 1: X=f32 k,  store bf16 head-split   kh
//   mode 2: X=f32 v,  store bf16 TRANSPOSED   vhT[(b*16+h)*64 + dh][S]
//   mode 3: X=bf16 z, store f32               out[gm][gn]
// One wave computes a 32x64 tile: 8 WMMAs per K-step against 6 fragment
// loads (2A + 4B) -> 1.33 MACs per load vs 1.0 for 32x32 blocking.
// ---------------------------------------------------------------------------
__global__ void proj_gemm_kernel(const float* __restrict__ Xf32,
                                 const unsigned short* __restrict__ Xbf,
                                 const unsigned short* __restrict__ WT,
                                 const float* __restrict__ bias,
                                 unsigned short* __restrict__ outHead,
                                 float* __restrict__ outF32,
                                 int mode) {
  int tid  = threadIdx.x;
  int lane = tid & 31;
  int wid  = (blockIdx.x * blockDim.x + tid) >> 5;   // 0..2047
  int mblk = (wid >> 4) * 32;                        // row0 (0..4064)
  int nblk = (wid & 15) * 64;                        // col0 (0..960)

  v8f acc[2][4] = {};
  for (int k0 = 0; k0 < DM; k0 += 32) {
    // prefetch next K tile (global_prefetch_b8 on gfx1250)
    if (k0 + 32 < DM) {
      if (mode == 3)
        __builtin_prefetch(Xbf + (mblk + (lane & 15)) * DM + k0 + 32, 0, 1);
      else
        __builtin_prefetch(Xf32 + (mblk + (lane & 15)) * DM + k0 + 32, 0, 1);
      __builtin_prefetch(WT + (nblk + lane) * DM + k0 + 32, 0, 1);
      __builtin_prefetch(WT + (nblk + 32 + lane) * DM + k0 + 32, 0, 1);
    }
    v16bf a0, a1;
    if (mode == 3) {
      a0 = load_a_bf16(Xbf + mblk * DM + k0, DM, lane);
      a1 = load_a_bf16(Xbf + (mblk + 16) * DM + k0, DM, lane);
    } else {
      a0 = load_a_f32(Xf32 + mblk * DM + k0, DM, lane);
      a1 = load_a_f32(Xf32 + (mblk + 16) * DM + k0, DM, lane);
    }
#pragma unroll
    for (int ni = 0; ni < 4; ++ni) {
      v16bf b = load_b_bf16(WT + (nblk + ni * 16) * DM + k0, DM, lane);
      acc[0][ni] = wmma_bf16(a0, b, acc[0][ni]);
      acc[1][ni] = wmma_bf16(a1, b, acc[1][ni]);
    }
  }

  int hh = lane >> 4, n = lane & 15;
#pragma unroll
  for (int mi = 0; mi < 2; ++mi) {
#pragma unroll
    for (int ni = 0; ni < 4; ++ni) {
      int gn = nblk + ni * 16 + n;
      float bv = bias[gn];
#pragma unroll
      for (int r = 0; r < 8; ++r) {
        int gm = mblk + mi * 16 + hh * 8 + r;   // D layout: m = r + 8*half
        float val = acc[mi][ni][r] + bv;
        if (mode == 3) {
          outF32[gm * DM + gn] = val;
        } else {
          int b = gm >> 11, s = gm & (SS - 1);
          int head = gn >> 6, dh = gn & (DH - 1);
          unsigned short bits = f32_to_bf16_bits(val);
          if (mode == 2)
            outHead[((b * NH + head) * DH + dh) * SS + s] = bits;      // V^T
          else
            outHead[((b * NH + head) * SS + s) * DH + dh] = bits;      // Q,K
        }
      }
    }
  }
}

// ---------------------------------------------------------------------------
// Kernel 3: attention. One WG = one (b,h) x 32 query rows, 8 waves.
// Scores for the full Sk=2048 live in LDS (32 x 2064 f32 = 258 KB — CDNA5's
// 320 KB/WGP LDS lets softmax + attn write + PV read the score tile without
// round-tripping HBM).
// ---------------------------------------------------------------------------
#define SC_LD 2064
#define SMEM_FLOATS (32 * SC_LD + 32 * 8 + 32)

__global__ void attn_kernel(const unsigned short* __restrict__ qh,
                            const unsigned short* __restrict__ kh,
                            const unsigned short* __restrict__ vhT,
                            const int* __restrict__ mask,
                            float* __restrict__ attn_out,
                            unsigned short* __restrict__ z) {
  extern __shared__ float smem_f[];
  float* sc      = smem_f;                 // [32][SC_LD]
  float* red     = sc + 32 * SC_LD;        // [32][8]
  float* rowstat = red + 32 * 8;           // [32]

  int tid  = threadIdx.x;
  int lane = tid & 31;
  int wv   = tid >> 5;                     // wave 0..7
  int bh   = blockIdx.x >> 6;              // 0..31
  int q0   = (blockIdx.x & 63) * 32;       // query tile base within S

  const unsigned short* qbase = qh  + (bh * SS + q0) * DH;
  const unsigned short* kbase = kh  + bh * SS * DH;
  const unsigned short* vbase = vhT + bh * DH * SS;

  // ---- phase 1: S = Q.K^T, masked + scaled, into LDS -------------------
  int hh = lane >> 4, nn = lane & 15;
  for (int mt = 0; mt < 2; ++mt) {
    const unsigned short* arow = qbase + mt * 16 * DH;
    v16bf a0 = load_a_bf16(arow + 0, DH, lane);
    v16bf a1 = load_a_bf16(arow + 32, DH, lane);
    for (int nt = 0; nt < 16; ++nt) {
      int col0 = wv * 256 + nt * 16;
      if (nt + 1 < 16)
        __builtin_prefetch(kbase + (col0 + 16 + nn) * DH, 0, 1);
      v8f acc = {};
      v16bf b0 = load_b_bf16(kbase + col0 * DH + 0, DH, lane);
      acc = wmma_bf16(a0, b0, acc);
      v16bf b1 = load_b_bf16(kbase + col0 * DH + 32, DH, lane);
      acc = wmma_bf16(a1, b1, acc);
      int col = col0 + nn;
#pragma unroll
      for (int r = 0; r < 8; ++r) {
        int row = mt * 16 + hh * 8 + r;
        float mval = (float)mask[(q0 + row) * SS + col];
        sc[row * SC_LD + col] = (acc[r] + mval * NEGC) * ISC;
      }
    }
  }
  __syncthreads();

  // ---- phase 2: softmax + write attn ----------------------------------
  int row = tid >> 3;       // 32 rows
  int seg = tid & 7;        // 8 threads per row, 256 cols each
  float* rp = sc + row * SC_LD + seg * 256;

  float mx = -3.4e38f;
  for (int j = 0; j < 256; ++j) mx = fmaxf(mx, rp[j]);
  red[row * 8 + seg] = mx;
  __syncthreads();
  if (tid < 32) {
    float m2 = red[tid * 8];
#pragma unroll
    for (int j = 1; j < 8; ++j) m2 = fmaxf(m2, red[tid * 8 + j]);
    rowstat[tid] = m2;
  }
  __syncthreads();
  mx = rowstat[row];

  float sum = 0.0f;
  for (int j = 0; j < 256; ++j) {
    float e = __expf(rp[j] - mx);
    rp[j] = e;
    sum += e;
  }
  red[row * 8 + seg] = sum;
  __syncthreads();
  if (tid < 32) {
    float s2 = red[tid * 8];
#pragma unroll
    for (int j = 1; j < 8; ++j) s2 += red[tid * 8 + j];
    rowstat[tid] = 1.0f / s2;
  }
  __syncthreads();
  float inv = rowstat[row];

  float* attn_row = attn_out + (bh * SS + q0 + row) * SS + seg * 256;
  for (int j = 0; j < 256; ++j) {
    float p = rp[j] * inv;
    rp[j] = p;            // keep normalized probs in LDS for PV
    attn_row[j] = p;      // mandatory attn output
  }
  __syncthreads();

  // ---- phase 3: Z = P.V  (P from LDS f32->bf16, V^T from global) ------
  {
    int mt = wv >> 2, nt = wv & 3;          // 2 x 4 tiles, one per wave
    v8f acc = {};
    const float* pbase = sc + mt * 16 * SC_LD;
    for (int k0 = 0; k0 < SS; k0 += 32) {
      if (k0 + 32 < SS)
        __builtin_prefetch(vbase + (nt * 16 + nn) * SS + k0 + 32, 0, 1);
      v16bf a = load_a_f32(pbase + k0, SC_LD, lane);
      v16bf b = load_b_bf16(vbase + (nt * 16) * SS + k0, SS, lane);
      acc = wmma_bf16(a, b, acc);
    }
    int b = bh >> 4, hd = bh & 15;
#pragma unroll
    for (int r = 0; r < 8; ++r) {
      int qs = q0 + mt * 16 + hh * 8 + r;
      int gm = b * SS + qs;
      int gn = hd * DH + nt * 16 + nn;
      z[gm * DM + gn] = f32_to_bf16_bits(acc[r]);
    }
  }
}

// ---------------------------------------------------------------------------
// launch
// ---------------------------------------------------------------------------
extern "C" void kernel_launch(void* const* d_in, const int* in_sizes, int n_in,
                              void* d_out, int out_size, void* d_ws, size_t ws_size,
                              hipStream_t stream) {
  (void)in_sizes; (void)n_in; (void)out_size; (void)ws_size;

  const float* q    = (const float*)d_in[0];
  const float* k    = (const float*)d_in[1];
  const float* v    = (const float*)d_in[2];
  const int*   mask = (const int*)  d_in[3];
  const float* Wq   = (const float*)d_in[4];
  const float* bq   = (const float*)d_in[5];
  const float* Wk   = (const float*)d_in[6];
  const float* bk   = (const float*)d_in[7];
  const float* Wv   = (const float*)d_in[8];
  const float* bv   = (const float*)d_in[9];
  const float* Wo   = (const float*)d_in[10];
  const float* bo   = (const float*)d_in[11];

  float* out_f32  = (float*)d_out;                       // [2,2048,1024]
  float* attn_f32 = out_f32 + (size_t)MROWS * DM;        // [2,16,2048,2048]

  // workspace carve (bf16 stored as ushort bits): 40 MB total
  char* ws = (char*)d_ws;
  const size_t MB = 1024 * 1024;
  unsigned short* WqT = (unsigned short*)(ws + 0 * MB);   // 2 MB each
  unsigned short* WkT = (unsigned short*)(ws + 2 * MB);
  unsigned short* WvT = (unsigned short*)(ws + 4 * MB);
  unsigned short* WoT = (unsigned short*)(ws + 6 * MB);
  unsigned short* qh  = (unsigned short*)(ws + 8 * MB);   // 8 MB each
  unsigned short* kh  = (unsigned short*)(ws + 16 * MB);
  unsigned short* vhT = (unsigned short*)(ws + 24 * MB);
  unsigned short* z   = (unsigned short*)(ws + 32 * MB);

  dim3 tb(32, 8), tg(32, 32);
  hipLaunchKernelGGL(transpose_cvt_kernel, tg, tb, 0, stream, Wq, WqT);
  hipLaunchKernelGGL(transpose_cvt_kernel, tg, tb, 0, stream, Wk, WkT);
  hipLaunchKernelGGL(transpose_cvt_kernel, tg, tb, 0, stream, Wv, WvT);
  hipLaunchKernelGGL(transpose_cvt_kernel, tg, tb, 0, stream, Wo, WoT);

  // projections: 2048 waves = 256 blocks x 256 threads (32x64 per wave)
  hipLaunchKernelGGL(proj_gemm_kernel, dim3(256), dim3(256), 0, stream,
                     q, (const unsigned short*)nullptr, WqT, bq, qh, (float*)nullptr, 0);
  hipLaunchKernelGGL(proj_gemm_kernel, dim3(256), dim3(256), 0, stream,
                     k, (const unsigned short*)nullptr, WkT, bk, kh, (float*)nullptr, 1);
  hipLaunchKernelGGL(proj_gemm_kernel, dim3(256), dim3(256), 0, stream,
                     v, (const unsigned short*)nullptr, WvT, bv, vhT, (float*)nullptr, 2);

  // attention: 32 (b,h) x 64 q-tiles = 2048 blocks, 258 KB dynamic LDS
  size_t smem = (size_t)SMEM_FLOATS * sizeof(float);
  hipLaunchKernelGGL(attn_kernel, dim3(2048), dim3(256), smem, stream,
                     qh, kh, vhT, mask, attn_f32, z);

  // output projection
  hipLaunchKernelGGL(proj_gemm_kernel, dim3(256), dim3(256), 0, stream,
                     (const float*)nullptr, z, WoT, bo,
                     (unsigned short*)nullptr, out_f32, 3);
}
